// OutsideLayer_29678224015893
// MI455X (gfx1250) — matrix operand order repaired and checked by reading, not compiled, not run on hardware
//
#include <hip/hip_runtime.h>
#include <hip/hip_bf16.h>
#include <math.h>
#include <stdint.h>

// ---------------------------------------------------------------------------
// Types for CDNA5 WMMA (gfx1250, wave32)
// ---------------------------------------------------------------------------
typedef __attribute__((ext_vector_type(16))) __bf16 v16bf;
typedef __attribute__((ext_vector_type(8)))  __bf16 v8bf;
typedef __attribute__((ext_vector_type(8)))  float  v8f;
typedef __attribute__((__vector_size__(16))) int    i32x4_;   // matches builtin proto

#define D_ 768
#define I_ 3072
#define C_ 128
#define B_ 2048
#define G_ 8
#define H_ 12
#define N_ (B_*G_)          // 16384 (batch*comb)

// GEMM epilogue flags
#define F_GELU    1
#define F_RESF32  2
#define F_RESBF16 4
#define F_OUTF32  8

// GEMM tiling
#define BM 256
#define BN 128
#define BK 32
#define LPAD 40            // LDS row stride in bf16 elems (20 banks -> conflict-free)

// CDNA5 async global->LDS copy (ASYNCcnt path), with portable fallback
#if defined(__HIP_DEVICE_COMPILE__) && defined(__gfx1250__) && \
    defined(__has_builtin) && __has_builtin(__builtin_amdgcn_global_load_async_to_lds_b128)
#define HAVE_ASYNC_LDS 1
#else
#define HAVE_ASYNC_LDS 0
#endif

__device__ __forceinline__ void stage16(const __bf16* g, __bf16* l) {
#if HAVE_ASYNC_LDS
  __builtin_amdgcn_global_load_async_to_lds_b128(
      (__attribute__((address_space(1))) i32x4_*)(uintptr_t)(g),
      (__attribute__((address_space(3))) i32x4_*)(uintptr_t)(l),
      0, 0);
#else
  *(v8bf*)l = *(const v8bf*)g;   // global_load_b128 + ds_store_b128
#endif
}

__device__ __forceinline__ void stage_join() {
#if HAVE_ASYNC_LDS
  #if __has_builtin(__builtin_amdgcn_s_wait_asynccnt)
  __builtin_amdgcn_s_wait_asynccnt(0);
  #else
  asm volatile("s_wait_asynccnt 0x0" ::: "memory");
  #endif
#endif
}

__device__ __forceinline__ float gelu_exact(float x) {
  return 0.5f * x * (1.0f + erff(x * 0.70710678118654752f));
}

union FragU { v16bf v; v8bf h[2]; };

// ---------------------------------------------------------------------------
// LDS-staged WMMA GEMM: Out[m,n] = epilogue( sum_k A[m,k]*W[n,k] + bias[n] )
// A: bf16, row stride aStride elems. W: bf16 (Nn, K) row-major. bias: f32.
// Block = 256 thr = 8 waves; block tile 256x128; wave tile 64x64 = 4x4 WMMA.
// A/B K-slabs staged to LDS via async copies; fragments via ds_load_b128.
// Requires M%256==0, Nn%128==0, K%32==0 (true for all uses here).
// ---------------------------------------------------------------------------
__global__ __launch_bounds__(256) void gemm_wmma(
    const __bf16* __restrict__ A, long aStride,
    const __bf16* __restrict__ W, const float* __restrict__ bias,
    void* __restrict__ Out, const void* __restrict__ Res,
    int M, int Nn, int K, int flags)
{
  __shared__ __align__(16) __bf16 smA[BM * LPAD];   // 20 KB
  __shared__ __align__(16) __bf16 smB[BN * LPAD];   // 10 KB

  const int tid  = threadIdx.x;
  const int lane = tid & 31;
  const int wid  = tid >> 5;
  const int wm   = wid & 3;                  // 4 wave-rows   (4 * 64 = 256)
  const int wn   = wid >> 2;                 // 2 wave-cols   (2 * 64 = 128)
  const int mBlock = blockIdx.x * BM;
  const int nBlock = blockIdx.y * BN;
  const int r  = lane & 15;
  const int hs = lane >> 4;

  v8f acc[4][4] = {};

  for (int kt = 0; kt < K; kt += BK) {
    __syncthreads();                          // prior iteration's reads done
    // Stage A slab: BM x 32 bf16 = 16 KB = 1024 x 16B chunks (4 per thread)
    #pragma unroll
    for (int q = 0; q < 4; ++q) {
      const int cid = q * 256 + tid;
      const int row = cid >> 2, c = cid & 3;  // 4 chunks per 64B row
      const __bf16* g = A + (size_t)(mBlock + row) * (size_t)aStride + kt + c * 8;
      stage16(g, &smA[row * LPAD + c * 8]);
    }
    // Stage B slab: BN x 32 bf16 = 8 KB = 512 x 16B chunks (2 per thread)
    #pragma unroll
    for (int q = 0; q < 2; ++q) {
      const int cid = q * 256 + tid;
      const int row = cid >> 2, c = cid & 3;
      const __bf16* g = W + (size_t)(nBlock + row) * (size_t)K + kt + c * 8;
      stage16(g, &smB[row * LPAD + c * 8]);
    }
    stage_join();                             // ASYNCcnt -> 0 (this wave's share)
    __syncthreads();                          // all waves' slabs visible

    // B fragments: contiguous k order (this code's choice): lane n=l&15, K = (l>>4)*16..+15
    FragU bfr[4];
    #pragma unroll
    for (int j = 0; j < 4; ++j) {
      const __bf16* p = &smB[(wn * 64 + j * 16 + r) * LPAD + hs * 16];
      bfr[j].h[0] = *(const v8bf*)(p);
      bfr[j].h[1] = *(const v8bf*)(p + 8);
    }
    // A fragments: contiguous k order (this code's choice): lane(l<16) K{0..15}, lane+16 K{16..31} (matches B)
    #pragma unroll
    for (int i = 0; i < 4; ++i) {
      FragU af;
      const __bf16* p = &smA[(wm * 64 + i * 16 + r) * LPAD + hs * 16];
      af.h[0] = *(const v8bf*)(p);
      af.h[1] = *(const v8bf*)(p + 8);
      #pragma unroll
      for (int j = 0; j < 4; ++j)
        acc[i][j] = __builtin_amdgcn_wmma_f32_16x16x32_bf16(
            false, af.v, false, bfr[j].v, (short)0, acc[i][j], false, false);
    }
  }

  // C/D layout: element (vgpr v, lane l) -> m_local = v + 8*(l>>4), n_local = l&15
  const int mW = mBlock + wm * 64;
  const int nW = nBlock + wn * 64;
  #pragma unroll
  for (int i = 0; i < 4; ++i) {
    #pragma unroll
    for (int j = 0; j < 4; ++j) {
      const int gn = nW + j * 16 + r;
      const float bn = bias ? bias[gn] : 0.0f;
      #pragma unroll
      for (int v = 0; v < 8; ++v) {
        const int gm = mW + i * 16 + v + 8 * hs;
        const size_t o = (size_t)gm * (size_t)Nn + gn;
        float x = acc[i][j][v] + bn;
        if (flags & F_GELU)    x = gelu_exact(x);
        if (flags & F_RESF32)  x += ((const float*)Res)[o];
        if (flags & F_RESBF16) x += (float)((const __bf16*)Res)[o];
        if (flags & F_OUTF32)  ((float*)Out)[o] = x;
        else                   ((__bf16*)Out)[o] = (__bf16)x;
      }
    }
  }
}

// ---------------------------------------------------------------------------
// f32 -> bf16 convert
// ---------------------------------------------------------------------------
__global__ void cvt_bf16(const float* __restrict__ src, __bf16* __restrict__ dst, size_t n) {
  size_t i = (size_t)blockIdx.x * 256 + threadIdx.x;
  if (i < n) dst[i] = (__bf16)src[i];
}

// ---------------------------------------------------------------------------
// Build kv_in = [child_l, child_r, parent] + role_emb, bf16   (N,3,D)
// ---------------------------------------------------------------------------
__global__ void build_x(const float* __restrict__ child, const float* __restrict__ parent,
                        const float* __restrict__ role, __bf16* __restrict__ XB) {
  size_t idx = (size_t)blockIdx.x * 256 + threadIdx.x;
  int d = (int)(idx % D_);
  size_t row = idx / D_;
  int rr = (int)(row % 3);
  size_t n = row / 3;
  float v = (rr < 2) ? child[(n*2 + rr)*D_ + d] : parent[(n / G_)*D_ + d];
  XB[idx] = (__bf16)(v + role[rr*D_ + d]);
}

// ---------------------------------------------------------------------------
// Attention: q_len=2, kv_len=3, H=12 heads of 64. One wave per (n,h).
// ---------------------------------------------------------------------------
__device__ __forceinline__ float wave_sum(float x) {
  #pragma unroll
  for (int o = 16; o > 0; o >>= 1) x += __shfl_xor(x, o, 32);
  return x;
}

__global__ __launch_bounds__(256) void attn_kernel(
    const __bf16* __restrict__ Q, const __bf16* __restrict__ Kb,
    const __bf16* __restrict__ V, __bf16* __restrict__ AO) {
  int gw = (int)((blockIdx.x * 256 + threadIdx.x) >> 5);
  int lane = threadIdx.x & 31;
  int n = gw / H_;
  int h = gw - n * H_;
  size_t colBase = (size_t)h * 64 + lane;

  float q[2][2], k[3][2], vv[3][2];
  #pragma unroll
  for (int qq = 0; qq < 2; ++qq) {
    size_t ro = ((size_t)(n*3 + qq)) * D_ + colBase;
    q[qq][0] = (float)Q[ro]; q[qq][1] = (float)Q[ro + 32];
  }
  #pragma unroll
  for (int rr = 0; rr < 3; ++rr) {
    size_t ro = ((size_t)(n*3 + rr)) * D_ + colBase;
    k[rr][0] = (float)Kb[ro]; k[rr][1] = (float)Kb[ro + 32];
    vv[rr][0] = (float)V[ro]; vv[rr][1] = (float)V[ro + 32];
  }
  float s[2][3];
  #pragma unroll
  for (int qq = 0; qq < 2; ++qq)
    #pragma unroll
    for (int rr = 0; rr < 3; ++rr)
      s[qq][rr] = wave_sum(q[qq][0]*k[rr][0] + q[qq][1]*k[rr][1]) * 0.125f; // 1/sqrt(64)

  #pragma unroll
  for (int qq = 0; qq < 2; ++qq) {
    float m = fmaxf(s[qq][0], fmaxf(s[qq][1], s[qq][2]));
    float e0 = __expf(s[qq][0]-m), e1 = __expf(s[qq][1]-m), e2 = __expf(s[qq][2]-m);
    float inv = 1.0f / (e0 + e1 + e2);
    float a0 = e0*inv, a1 = e1*inv, a2 = e2*inv;
    float o0 = a0*vv[0][0] + a1*vv[1][0] + a2*vv[2][0];
    float o1 = a0*vv[0][1] + a1*vv[1][1] + a2*vv[2][1];
    size_t oro = ((size_t)(n*2 + qq)) * D_ + colBase;
    AO[oro] = (__bf16)o0; AO[oro + 32] = (__bf16)o1;
  }
}

// ---------------------------------------------------------------------------
// Row LayerNorm over D=768. One block (256 thr) per row.
// ---------------------------------------------------------------------------
__global__ __launch_bounds__(256) void ln_kernel(
    const __bf16* __restrict__ X, const float* __restrict__ g,
    const float* __restrict__ be, void* __restrict__ out, int outF32) {
  __shared__ float sA[256], sB[256];
  const size_t row = blockIdx.x;
  const __bf16* xr = X + row * D_;
  float a = 0.f, b = 0.f;
  for (int d = threadIdx.x; d < D_; d += 256) { float x = (float)xr[d]; a += x; b += x*x; }
  sA[threadIdx.x] = a; sB[threadIdx.x] = b; __syncthreads();
  for (int s = 128; s > 0; s >>= 1) {
    if ((int)threadIdx.x < s) { sA[threadIdx.x] += sA[threadIdx.x+s]; sB[threadIdx.x] += sB[threadIdx.x+s]; }
    __syncthreads();
  }
  float mean = sA[0] * (1.0f / D_);
  float var  = sB[0] * (1.0f / D_) - mean * mean;
  float inv  = rsqrtf(var + 1e-5f);
  for (int d = threadIdx.x; d < D_; d += 256) {
    float y = ((float)xr[d] - mean) * inv * g[d] + be[d];
    if (outF32) ((float*)out)[row * D_ + d] = y;
    else        ((__bf16*)out)[row * D_ + d] = (__bf16)y;
  }
}

// ---------------------------------------------------------------------------
// Final scoring: per (b,g): dot(Pc[b], Rc/Lc[b,g]) / sqrt(C) + scores
// ---------------------------------------------------------------------------
__global__ void score_kernel(const float* __restrict__ Pc, const float* __restrict__ Lc,
                             const float* __restrict__ Rc, const float* __restrict__ ps,
                             const float* __restrict__ cs, float* __restrict__ out) {
  int i = blockIdx.x * 256 + threadIdx.x;   // 0..N-1
  int b = i / G_;
  const float* p = Pc + (size_t)b * C_;
  const float* l = Lc + (size_t)i * C_;
  const float* r = Rc + (size_t)i * C_;
  float aL = 0.f, aR = 0.f;
  for (int c = 0; c < C_; ++c) { float pv = p[c]; aL += pv * r[c]; aR += pv * l[c]; }
  const float inv = 0.088388347648318447f;  // 1/sqrt(128)
  float pb = ps[b];
  out[(size_t)i*2 + 0] = aL * inv + pb + cs[(size_t)i*2 + 1];
  out[(size_t)i*2 + 1] = aR * inv + pb + cs[(size_t)i*2 + 0];
}

// ---------------------------------------------------------------------------
// Host launcher
// ---------------------------------------------------------------------------
extern "C" void kernel_launch(void* const* d_in, const int* in_sizes, int n_in,
                              void* d_out, int out_size, void* d_ws, size_t ws_size,
                              hipStream_t stream) {
  const float* parent = (const float*)d_in[0];
  const float* child  = (const float*)d_in[1];
  const float* pscore = (const float*)d_in[2];
  const float* cscore = (const float*)d_in[3];
  const float* role   = (const float*)d_in[4];
  const float* Wq = (const float*)d_in[5];  const float* bq = (const float*)d_in[6];
  const float* Wk = (const float*)d_in[7];  const float* bk = (const float*)d_in[8];
  const float* Wv = (const float*)d_in[9];  const float* bv = (const float*)d_in[10];
  const float* Wo = (const float*)d_in[11]; const float* bo = (const float*)d_in[12];
  const float* g1 = (const float*)d_in[13]; const float* be1 = (const float*)d_in[14];
  const float* g2 = (const float*)d_in[15]; const float* be2 = (const float*)d_in[16];
  const float* Wm1 = (const float*)d_in[17]; const float* bm1 = (const float*)d_in[18];
  const float* Wm2 = (const float*)d_in[19]; const float* bm2 = (const float*)d_in[20];
  const float* Wl1 = (const float*)d_in[21]; const float* bl1 = (const float*)d_in[22];
  const float* Wl2 = (const float*)d_in[23]; const float* bl2 = (const float*)d_in[24];
  const float* Wr1 = (const float*)d_in[25]; const float* br1 = (const float*)d_in[26];
  const float* Wr2 = (const float*)d_in[27]; const float* br2 = (const float*)d_in[28];
  const float* Wp1 = (const float*)d_in[29]; const float* bp1 = (const float*)d_in[30];
  const float* Wp2 = (const float*)d_in[31]; const float* bp2 = (const float*)d_in[32];

  char* base = (char*)d_ws;
  size_t off = 0;
  auto take = [&](size_t bytes) -> void* {
    void* p = base + off;
    off += (bytes + 255) & ~(size_t)255;
    return p;
  };

  // bf16 weights
  __bf16* wqB  = (__bf16*)take((size_t)D_*D_*2);
  __bf16* wkB  = (__bf16*)take((size_t)D_*D_*2);
  __bf16* wvB  = (__bf16*)take((size_t)D_*D_*2);
  __bf16* woB  = (__bf16*)take((size_t)D_*D_*2);
  __bf16* wm1B = (__bf16*)take((size_t)I_*D_*2);
  __bf16* wm2B = (__bf16*)take((size_t)D_*I_*2);
  __bf16* wl1B = (__bf16*)take((size_t)D_*D_*2);
  __bf16* wl2B = (__bf16*)take((size_t)C_*D_*2);
  __bf16* wr1B = (__bf16*)take((size_t)D_*D_*2);
  __bf16* wr2B = (__bf16*)take((size_t)C_*D_*2);
  __bf16* wp1B = (__bf16*)take((size_t)D_*D_*2);
  __bf16* wp2B = (__bf16*)take((size_t)C_*D_*2);
  // activations
  __bf16* XB  = (__bf16*)take((size_t)N_*3*D_*2);   // kv_in bf16
  __bf16* CB  = (__bf16*)take((size_t)N_*2*D_*2);   // child bf16
  __bf16* PB  = (__bf16*)take((size_t)B_*D_*2);     // parent bf16
  __bf16* QB  = (__bf16*)take((size_t)N_*3*D_*2);
  __bf16* KB  = (__bf16*)take((size_t)N_*3*D_*2);
  __bf16* VB  = (__bf16*)take((size_t)N_*3*D_*2);
  __bf16* AB  = (__bf16*)take((size_t)N_*2*D_*2);   // attention out
  __bf16* H1B = (__bf16*)take((size_t)N_*2*D_*2);   // pre-LN1
  __bf16* S1B = (__bf16*)take((size_t)N_*2*D_*2);   // post-LN1
  __bf16* MB  = (__bf16*)take((size_t)N_*2*I_*2);   // gelu(mid)
  __bf16* H2B = (__bf16*)take((size_t)N_*2*D_*2);   // pre-LN2
  __bf16* LmB = (__bf16*)take((size_t)N_*D_*2);
  __bf16* RmB = (__bf16*)take((size_t)N_*D_*2);
  __bf16* PmB = (__bf16*)take((size_t)B_*D_*2);
  float*  LcF = (float*)take((size_t)N_*C_*4);
  float*  RcF = (float*)take((size_t)N_*C_*4);
  float*  PcF = (float*)take((size_t)B_*C_*4);

  float* outScores = (float*)d_out;                  // (B,G,2)
  float* outSrc    = (float*)d_out + (size_t)N_*2;   // (B,G,2,D)

  auto cvt = [&](const float* s, __bf16* d, size_t n) {
    cvt_bf16<<<dim3((unsigned)((n + 255) / 256)), 256, 0, stream>>>(s, d, n);
  };

  // --- stage 0: conversions / input build -------------------------------
  cvt(Wq, wqB, (size_t)D_*D_);   cvt(Wk, wkB, (size_t)D_*D_);
  cvt(Wv, wvB, (size_t)D_*D_);   cvt(Wo, woB, (size_t)D_*D_);
  cvt(Wm1, wm1B, (size_t)I_*D_); cvt(Wm2, wm2B, (size_t)D_*I_);
  cvt(Wl1, wl1B, (size_t)D_*D_); cvt(Wl2, wl2B, (size_t)C_*D_);
  cvt(Wr1, wr1B, (size_t)D_*D_); cvt(Wr2, wr2B, (size_t)C_*D_);
  cvt(Wp1, wp1B, (size_t)D_*D_); cvt(Wp2, wp2B, (size_t)C_*D_);
  cvt(child, CB, (size_t)N_*2*D_);
  cvt(parent, PB, (size_t)B_*D_);
  build_x<<<dim3((unsigned)(((size_t)N_*3*D_) / 256)), 256, 0, stream>>>(child, parent, role, XB);

  dim3 blk(256);
  const int M3 = N_*3, M2 = N_*2;

  // --- stage 1: QKV projections (WMMA) ----------------------------------
  gemm_wmma<<<dim3(M3/BM, D_/BN), blk, 0, stream>>>(XB, D_, wqB, bq, QB, nullptr, M3, D_, D_, 0);
  gemm_wmma<<<dim3(M3/BM, D_/BN), blk, 0, stream>>>(XB, D_, wkB, bk, KB, nullptr, M3, D_, D_, 0);
  gemm_wmma<<<dim3(M3/BM, D_/BN), blk, 0, stream>>>(XB, D_, wvB, bv, VB, nullptr, M3, D_, D_, 0);

  // --- stage 2: tiny attention (2x3 per head), wave per (n,h) -----------
  attn_kernel<<<dim3((N_*H_)/8), blk, 0, stream>>>(QB, KB, VB, AB);

  // --- stage 3: Wo projection + residual, LN1 ---------------------------
  gemm_wmma<<<dim3(M2/BM, D_/BN), blk, 0, stream>>>(AB, D_, woB, bo, H1B, child, M2, D_, D_, F_RESF32);
  ln_kernel<<<dim3(M2), blk, 0, stream>>>(H1B, g1, be1, S1B, 0);

  // --- stage 4: MLP + residual, LN2 -> output src -----------------------
  gemm_wmma<<<dim3(M2/BM, I_/BN), blk, 0, stream>>>(S1B, D_, wm1B, bm1, MB, nullptr, M2, I_, D_, F_GELU);
  gemm_wmma<<<dim3(M2/BM, D_/BN), blk, 0, stream>>>(MB, I_, wm2B, bm2, H2B, S1B, M2, D_, I_, F_RESBF16);
  ln_kernel<<<dim3(M2), blk, 0, stream>>>(H2B, g2, be2, outSrc, 1);

  // --- stage 5: scoring branches ----------------------------------------
  gemm_wmma<<<dim3(N_/BM, D_/BN), blk, 0, stream>>>(CB,      2*D_, wl1B, bl1, LmB, nullptr, N_, D_, D_, F_GELU);
  gemm_wmma<<<dim3(N_/BM, D_/BN), blk, 0, stream>>>(CB + D_, 2*D_, wr1B, br1, RmB, nullptr, N_, D_, D_, F_GELU);
  gemm_wmma<<<dim3(B_/BM, D_/BN), blk, 0, stream>>>(PB,      D_,   wp1B, bp1, PmB, nullptr, B_, D_, D_, F_GELU);
  gemm_wmma<<<dim3(N_/BM, C_/BN), blk, 0, stream>>>(LmB, D_, wl2B, bl2, LcF, nullptr, N_, C_, D_, F_OUTF32);
  gemm_wmma<<<dim3(N_/BM, C_/BN), blk, 0, stream>>>(RmB, D_, wr2B, br2, RcF, nullptr, N_, C_, D_, F_OUTF32);
  gemm_wmma<<<dim3(B_/BM, C_/BN), blk, 0, stream>>>(PmB, D_, wp2B, bp2, PcF, nullptr, B_, C_, D_, F_OUTF32);

  // --- stage 6: outside scores ------------------------------------------
  score_kernel<<<dim3(N_/256), blk, 0, stream>>>(PcF, LcF, RcF, pscore, cscore, outScores);

  (void)in_sizes; (void)n_in; (void)out_size; (void)ws_size;
}
